// Discriminator_29446295781422
// MI455X (gfx1250) — compile-verified, matrix operand order checked by reading, not run on hardware
//
#include <hip/hip_runtime.h>

typedef __attribute__((ext_vector_type(16))) __bf16 v16bf;
typedef __attribute__((ext_vector_type(8)))  __bf16 v8bf;
typedef __attribute__((ext_vector_type(8)))  float  v8f;
typedef __attribute__((ext_vector_type(4)))  unsigned u32x4;
typedef __attribute__((ext_vector_type(8)))  int      i32x8;
typedef __attribute__((ext_vector_type(4)))  int      i32x4;

#define GB 128   // batch
#define GT 512   // time steps
#define GI 256   // input size
#define GH 256   // hidden size
#define G3H 768

static __device__ __forceinline__ v8f wmma_bf16(v16bf a, v16bf b, v8f c) {
  return __builtin_amdgcn_wmma_f32_16x16x32_bf16(false, a, false, b, (short)0, c,
                                                 false, false);
}

// Load one 16x32 bf16 fragment (ISA 7.12.2 layout) from row-major tile.
static __device__ __forceinline__ v16bf frag16(const __bf16* base, int ld) {
  const int lane = threadIdx.x & 31;
  const int r = lane & 15, g = lane >> 4;
  const __bf16* p0 = base + (size_t)r * ld + g * 8;
  v8bf lo = *reinterpret_cast<const v8bf*>(p0);
  v8bf hi = *reinterpret_cast<const v8bf*>(p0 + 16);
  return __builtin_shufflevector(lo, hi, 0,1,2,3,4,5,6,7,8,9,10,11,12,13,14,15);
}

static __device__ __forceinline__ float sigm(float x) {
  return 1.0f / (1.0f + __expf(-x));
}

// Issue one TDM DMA: contiguous `elems` f32 from `gaddr` into LDS offset
// `lds_off`. 2-D descriptor (groups 2/3 zero). Wave-uniform; caller guards
// to a single wave. Tracked by TENSORcnt.
// clang-23 toolchain: 6-arg builtin (g0 x4, g1 x8, g2 x4, g3 x4, x8, cpol).
static __device__ __forceinline__ void tdm_load(unsigned lds_off,
                                                unsigned long long gaddr,
                                                unsigned elems) {
  u32x4 g0;
  g0[0] = 1u;                                   // count=1, user descriptor
  g0[1] = lds_off;                              // lds_addr [63:32]
  g0[2] = (unsigned)gaddr;                      // global_addr [95:64]
  g0[3] = (unsigned)((gaddr >> 32) & 0x1FFFFFFull) | (2u << 30); // addr hi | type=2
  i32x8 g1;
  g1[0] = (int)(2u << 16);                      // data_size=4B, no multicast/pad
  g1[1] = (int)((elems & 0xFFFFu) << 16);       // tensor_dim0[15:0] at bits 63:48
  g1[2] = (int)(((elems >> 16) & 0xFFFFu) | (1u << 16)); // dim0 hi | tensor_dim1=1
  g1[3] = (int)((elems & 0xFFFFu) << 16);       // tile_dim0 at bits 127:112
  g1[4] = 1;                                    // tile_dim1=1, tile_dim2=0
  g1[5] = (int)elems;                           // tensor_dim0_stride lo
  g1[6] = 0;
  g1[7] = 0;
  i32x4 gz  = {0, 0, 0, 0};
  i32x8 gz8 = {0, 0, 0, 0, 0, 0, 0, 0};
  __builtin_amdgcn_tensor_load_to_lds(g0, g1, gz, gz, gz8, 0);
}

// ---------------------------------------------------------------- f32 -> bf16
__global__ void f2bf(const float* __restrict__ src, __bf16* __restrict__ dst, int n) {
  int i = blockIdx.x * blockDim.x + threadIdx.x;
  if (i < n) dst[i] = (__bf16)src[i];
}

// ------------------------------------------------- x-projection GEMM (WMMA)
// out[t][b][n] = sum_k A[b*T+t][k] * W[n][k] + bias[n]
// K outer, 4 live accumulators, double-buffered B frags so loads overlap WMMA.
__global__ void __launch_bounds__(256) xproj_gemm(
    const __bf16* __restrict__ A, const __bf16* __restrict__ W,
    const float* __restrict__ bias, float* __restrict__ out) {
  const int wave = (blockIdx.x * blockDim.x + threadIdx.x) >> 5;
  const int mt = wave / 12;          // 4096 m-tiles of 16 rows
  const int ng = wave % 12;          // 4 n-tiles per wave
  const int lane = threadIdx.x & 31, r = lane & 15, g = lane >> 4;

  const __bf16* Abase = A + (size_t)mt * 16 * GI;
  v16bf a[8];
#pragma unroll
  for (int kf = 0; kf < 8; ++kf) a[kf] = frag16(Abase + kf * 32, GI);

  v8f acc[4];
#pragma unroll
  for (int i = 0; i < 4; ++i) {
    const float bv = bias[(ng * 4 + i) * 16 + r];
    acc[i] = (v8f){bv, bv, bv, bv, bv, bv, bv, bv};
  }

  v16bf bcur[4], bnxt[4];
#pragma unroll
  for (int i = 0; i < 4; ++i)
    bcur[i] = frag16(W + (size_t)((ng * 4 + i) * 16) * GI, GI);

#pragma unroll
  for (int kf = 0; kf < 8; ++kf) {
    if (kf < 7) {
#pragma unroll
      for (int i = 0; i < 4; ++i)
        bnxt[i] = frag16(W + (size_t)((ng * 4 + i) * 16) * GI + (kf + 1) * 32, GI);
    }
#pragma unroll
    for (int i = 0; i < 4; ++i) acc[i] = wmma_bf16(a[kf], bcur[i], acc[i]);
#pragma unroll
    for (int i = 0; i < 4; ++i) bcur[i] = bnxt[i];
  }

#pragma unroll
  for (int i = 0; i < 4; ++i) {
    const int n0 = (ng * 4 + i) * 16;
#pragma unroll
    for (int v = 0; v < 8; ++v) {
      const int m = mt * 16 + v + 8 * g;    // global row = b*T + t
      const int bi = m >> 9;                // / T
      const int t  = m & (GT - 1);
      out[((size_t)t * GB + bi) * G3H + n0 + r] = acc[i][v];
    }
  }
}

// ---------------------------------------------------------------- GRU scan
// 8 blocks (batch tiles of 16) x 512 threads (16 waves).
// Wave w owns hidden cols [16w,16w+16); Whh slices register-stationary.
// Step t+1's 48 KB x-proj tile is DMA'd by the Tensor Data Mover into the
// other LDS buffer while step t's WMMA chain runs.
__global__ void __launch_bounds__(512) gru_scan(
    const float* __restrict__ xproj,   // [T][128][768] f32 (has bih added)
    const __bf16* __restrict__ Whh,    // [768][256] bf16
    const float* __restrict__ bhh,     // [768]
    __bf16* __restrict__ out_seq,      // [B][T][256] bf16 or nullptr
    float* __restrict__ hT)            // [B][256] f32 or nullptr
{
  __shared__ __align__(16) float  xt[2][16][G3H];      // 96 KB, double buffered
  __shared__ __align__(16) __bf16 hb[2][16][GH + 8];   // padded rows
  const int tid = threadIdx.x;
  const int w = tid >> 5, lane = tid & 31, r = lane & 15, g = lane >> 4;
  const int b0 = blockIdx.x * 16;

  // h0 = 0
  for (int i = tid; i < 16 * (GH + 8); i += 512) (&hb[0][0][0])[i] = (__bf16)0.0f;

  // Register-stationary recurrent weights: 3 gates x 8 K-fragments.
  v16bf bw[24];
#pragma unroll
  for (int gg = 0; gg < 3; ++gg) {
    const int row0 = gg * GH + w * 16;
#pragma unroll
    for (int kf = 0; kf < 8; ++kf)
      bw[gg * 8 + kf] = frag16(Whh + (size_t)row0 * GH + kf * 32, GH);
  }
  const float br = bhh[w * 16 + r];
  const float bz = bhh[GH + w * 16 + r];
  const float bn = bhh[2 * GH + w * 16 + r];
  __syncthreads();

  const unsigned ldsA = (unsigned)(unsigned long long)(&xt[0][0][0]);
  const unsigned ldsB = (unsigned)(unsigned long long)(&xt[1][0][0]);

  // TDM prefetch of tile t=0 (wave 0 only; uniform branch).
  if (w == 0)
    tdm_load(ldsA, (unsigned long long)(xproj + (size_t)b0 * G3H), 16 * G3H);

  int p = 0, q = 0;
  for (int t = 0; t < GT; ++t) {
    if (w == 0) __builtin_amdgcn_s_wait_tensorcnt(0);
    __syncthreads();   // xt[q] DMA complete; hb[p] writes from prev step visible

    if (t + 1 < GT && w == 0) {   // DMA next tile, overlapped with WMMA below
      tdm_load(q ? ldsA : ldsB,
               (unsigned long long)(xproj + ((size_t)(t + 1) * GB + b0) * G3H),
               16 * G3H);
    }

    // Recurrent GEMM: [16 x 256] h-tile times the wave's 3 gate slices.
    v8f accr = {}, accz = {}, accn = {};
#pragma unroll
    for (int kf = 0; kf < 8; ++kf) {
      v16bf a = frag16(&hb[p][0][0] + kf * 32, GH + 8);
      accr = wmma_bf16(a, bw[kf],      accr);
      accz = wmma_bf16(a, bw[8 + kf],  accz);
      accn = wmma_bf16(a, bw[16 + kf], accn);
    }

    // Gate math per C-element: m = v + 8g, col = 16w + r.
#pragma unroll
    for (int v = 0; v < 8; ++v) {
      const int m = v + 8 * g;
      const int col = w * 16 + r;
      const float xr = xt[q][m][col];
      const float xz = xt[q][m][GH + col];
      const float xn = xt[q][m][2 * GH + col];
      const float rg = sigm(xr + accr[v] + br);
      const float zg = sigm(xz + accz[v] + bz);
      const float ngt = tanhf(xn + rg * (accn[v] + bn));
      const float hold = (float)hb[p][m][col];
      const float hnew = (1.0f - zg) * ngt + zg * hold;
      hb[p ^ 1][m][col] = (__bf16)hnew;
      if (out_seq)
        out_seq[((size_t)(b0 + m) * GT + t) * GH + col] = (__bf16)hnew;
      if (hT && t == GT - 1)
        hT[(size_t)(b0 + m) * GH + col] = hnew;
    }
    p ^= 1;
    q ^= 1;
  }
}

// -------------------------------------------------------------- final head
__global__ void head(const float* __restrict__ hT, const float* __restrict__ Wfc,
                     const float* __restrict__ bfc, float* __restrict__ y) {
  const int b = threadIdx.x;
  float s = bfc[0];
  for (int j = 0; j < GH; ++j) s += hT[(size_t)b * GH + j] * Wfc[j];
  y[b] = 1.0f / (1.0f + __expf(-s));
}

// ---------------------------------------------------------------- launcher
extern "C" void kernel_launch(void* const* d_in, const int* in_sizes, int n_in,
                              void* d_out, int out_size, void* d_ws, size_t ws_size,
                              hipStream_t stream) {
  const float* x    = (const float*)d_in[0];
  const float* Wih0 = (const float*)d_in[1];
  const float* Whh0 = (const float*)d_in[2];
  const float* bih0 = (const float*)d_in[3];
  const float* bhh0 = (const float*)d_in[4];
  const float* Wih1 = (const float*)d_in[5];
  const float* Whh1 = (const float*)d_in[6];
  const float* bih1 = (const float*)d_in[7];
  const float* bhh1 = (const float*)d_in[8];
  const float* Wfc  = (const float*)d_in[9];
  const float* bfc  = (const float*)d_in[10];

  char* ws = (char*)d_ws;
  size_t off = 0;
  auto carve = [&](size_t bytes) {
    void* p = ws + off;
    off = (off + bytes + 255) & ~(size_t)255;
    return p;
  };
  __bf16* xb    = (__bf16*)carve((size_t)GB * GT * GI * 2);   // 32 MB
  __bf16* W0ib  = (__bf16*)carve((size_t)G3H * GI * 2);
  __bf16* W0hb  = (__bf16*)carve((size_t)G3H * GH * 2);
  __bf16* W1ib  = (__bf16*)carve((size_t)G3H * GH * 2);
  __bf16* W1hb  = (__bf16*)carve((size_t)G3H * GH * 2);
  float*  xproj = (float*)carve((size_t)GT * GB * G3H * 4);   // 201 MB (reused)
  __bf16* out0b = (__bf16*)carve((size_t)GB * GT * GH * 2);   // 32 MB
  float*  hT    = (float*)carve((size_t)GB * GH * 4);

  const int nx = GB * GT * GI;
  const int nw = G3H * GI;
  f2bf<<<(nx + 255) / 256, 256, 0, stream>>>(x,    xb,   nx);
  f2bf<<<(nw + 255) / 256, 256, 0, stream>>>(Wih0, W0ib, nw);
  f2bf<<<(nw + 255) / 256, 256, 0, stream>>>(Whh0, W0hb, nw);
  f2bf<<<(nw + 255) / 256, 256, 0, stream>>>(Wih1, W1ib, nw);
  f2bf<<<(nw + 255) / 256, 256, 0, stream>>>(Whh1, W1hb, nw);

  // Layer 0: x-projection GEMM, then batch-parallel scan.
  xproj_gemm<<<6144, 256, 0, stream>>>(xb, W0ib, bih0, xproj);
  gru_scan<<<GB / 16, 512, 0, stream>>>(xproj, W0hb, bhh0, out0b, nullptr);

  // Layer 1: projection of layer-0 outputs, then scan (keep only h_T).
  xproj_gemm<<<6144, 256, 0, stream>>>(out0b, W1ib, bih1, xproj);
  gru_scan<<<GB / 16, 512, 0, stream>>>(xproj, W1hb, bhh1, nullptr, hT);

  head<<<1, 128, 0, stream>>>(hT, Wfc, bfc, (float*)d_out);
}
